// CausalSelfAttention_84859963834598
// MI455X (gfx1250) — matrix operand order repaired and checked by reading, not compiled, not run on hardware
//
#include <hip/hip_runtime.h>

// ---------------------------------------------------------------------------
// CDNA5 (gfx1250, wave32) fused causal sliding-window attention pipeline.
// All matmuls run on v_wmma_f32_16x16x32_bf16 (f32 accumulate).
//   1) convert x -> bf16, pack [Wq|Wk|Wv] -> bf16, Wproj -> bf16
//   2) blocked GEMM (TDM A-tile staging + LDS B^T staging): qkv = x @ Wqkv
//   3) epilogue: RoPE + RMSnorm on q/k, sigmoid-gate + ve on v; writes
//      Q[B,H,T,64] row-major, K padded [B,KV,TPAD,64], V transposed+padded
//      [B,KV,64,TPAD] so every attention fragment load is contiguous b128s.
//   4) flash attention per 16-query tile, window 512 (WMMA, no clamps)
//   5) blocked GEMM: out = Y @ Wproj
// ---------------------------------------------------------------------------

typedef __bf16 bf16;
typedef __attribute__((ext_vector_type(16))) __bf16 v16bf;
typedef __attribute__((ext_vector_type(8)))  float  v8f;
typedef __attribute__((ext_vector_type(4)))  unsigned int u32x4;
typedef __attribute__((ext_vector_type(8)))  int i32x8;
typedef __attribute__((ext_vector_type(4)))  int i32x4;

#define N_HEAD 16
#define N_KV   4
#define WINDOW 512
#define N_EMBD 1024
#define BB     2
#define TT     2048
#define TPAD   (WINDOW + TT + 32)   // 2592: front pad 512, back pad 32

#if defined(__HIP_DEVICE_COMPILE__) && __has_builtin(__builtin_amdgcn_tensor_load_to_lds)
#define USE_TDM 1
#else
#define USE_TDM 0
#endif

union Frag16 { v16bf v; u32x4 q[2]; };

__device__ __forceinline__ v8f wmma_bf16(v16bf a, v16bf b, v8f c) {
  return __builtin_amdgcn_wmma_f32_16x16x32_bf16(false, a, false, b,
                                                 (short)0, c, false, false);
}

// A-fragment 16x32 (MxK) bf16 row-major, ld elements. Per-lane data is two
// contiguous 16B runs: [m*ld + 8*half .. +8) and [m*ld + 16 + 8*half .. +8).
__device__ __forceinline__ v16bf load_frag_a16(const bf16* p, int ld) {
  const int lane = threadIdx.x & 31;
  const int m = lane & 15, half = lane >> 4;
  const bf16* r = p + (size_t)m * ld + 8 * half;
  Frag16 f;
  f.q[0] = *(const u32x4*)r;
  f.q[1] = *(const u32x4*)(r + 16);
  return f.v;
}

// B-fragment 32x16 (KxN) where memory is K-major per column: element (k,n) at
// p[n*ld + k]. Per-lane (n = lane&15) data is two 16B runs:
// k = 8*half .. +8 and k = 16 + 8*half .. +8.
__device__ __forceinline__ v16bf load_frag_bk(const bf16* p, int ld) {
  const int lane = threadIdx.x & 31;
  const int n = lane & 15, half = lane >> 4;
  const bf16* r = p + (size_t)n * ld + 8 * half;
  Frag16 f;
  f.q[0] = *(const u32x4*)r;
  f.q[1] = *(const u32x4*)(r + 16);
  return f.v;
}

// ---------------------------------------------------------------------------
// Conversions / packing
// ---------------------------------------------------------------------------
__global__ void f32_to_bf16_kernel(const float* __restrict__ src,
                                   bf16* __restrict__ dst, int n4) {
  int i = blockIdx.x * blockDim.x + threadIdx.x;
  if (i >= n4) return;
  const float4 f = ((const float4*)src)[i];
  union { bf16 h[4]; unsigned long long u; } o;
  o.h[0] = (bf16)f.x; o.h[1] = (bf16)f.y; o.h[2] = (bf16)f.z; o.h[3] = (bf16)f.w;
  ((unsigned long long*)dst)[i] = o.u;
}

__global__ void pack_wqkv_kernel(const float* __restrict__ wq,
                                 const float* __restrict__ wk,
                                 const float* __restrict__ wv,
                                 bf16* __restrict__ dst) {
  int i = blockIdx.x * blockDim.x + threadIdx.x;  // over 1024*1536
  if (i >= N_EMBD * 1536) return;
  int k = i / 1536, n = i % 1536;
  float v;
  if (n < 1024)      v = wq[(size_t)k * 1024 + n];
  else if (n < 1280) v = wk[(size_t)k * 256 + (n - 1024)];
  else               v = wv[(size_t)k * 256 + (n - 1280)];
  dst[i] = (bf16)v;
}

// ---------------------------------------------------------------------------
// Blocked bf16 GEMM: C[M,N] (f32) = A[M,K] @ B[K,N].
// Block = 8 waves = 128(M) x 64(N) of C; per k-step of 32:
//   - A tile 128x32 staged to LDS by the Tensor Data Mover (wave 0 issues,
//     s_wait_tensorcnt, block barrier); cooperative fallback without TDM.
//   - B tile 32x64 staged transposed (K-major) into LDS cooperatively.
//   - each wave: 1 A-frag (reused 4x) x 4 B-frags -> 4 WMMAs.
// ---------------------------------------------------------------------------
#define BT_STRIDE 40  // padded K-major stride for btile (16B-aligned reads)

__global__ void __launch_bounds__(256)
gemm_bf16_kernel(const bf16* __restrict__ A, const bf16* __restrict__ Bm,
                 float* __restrict__ C, int M, int N, int K) {
  __shared__ bf16 atile[128 * 32];
  __shared__ bf16 btile[64 * BT_STRIDE];
  const int tid = threadIdx.x;
  const int warp = tid >> 5, lane = tid & 31;
  const int ntb = N >> 6;
  const int bm = blockIdx.x / ntb, bn = blockIdx.x % ntb;
  const int half = lane >> 4, nn = lane & 15;
  v8f acc0 = {}, acc1 = {}, acc2 = {}, acc3 = {};

  for (int k0 = 0; k0 < K; k0 += 32) {
#if USE_TDM
    if (warp == 0) {
      const unsigned long long ga =
          (unsigned long long)(size_t)(A + ((size_t)bm * 128) * K + k0);
      const unsigned lds = (unsigned)(size_t)(void*)atile;
      u32x4 g0;
      g0[0] = 1u;                                   // count=1, user desc
      g0[1] = lds;                                  // lds_addr
      g0[2] = (unsigned)ga;                         // global_addr[31:0]
      g0[3] = ((unsigned)(ga >> 32) & 0x01FFFFFFu)  // global_addr[56:32]
              | (2u << 30);                         // type=2 (image)
      i32x8 g1;
      g1[0] = (int)(1u << 16);                      // data_size=1 (2 bytes)
      g1[1] = (int)(((unsigned)K & 0xFFFFu) << 16); // tensor_dim0[15:0]
      g1[2] = (int)((((unsigned)K >> 16) & 0xFFFFu)
              | (((unsigned)M & 0xFFFFu) << 16));   // dim0[31:16] | dim1[15:0]
      g1[3] = (int)((((unsigned)M >> 16) & 0xFFFFu)
              | (32u << 16));                       // dim1[31:16] | tile_dim0=32
      g1[4] = 128;                                  // tile_dim1=128, tile_dim2=0
      g1[5] = K;                                    // tensor_dim0_stride[31:0]
      g1[6] = 0;                                    // stride hi | dim1_stride lo
      g1[7] = 0;
      i32x4 z4 = {0, 0, 0, 0};
      i32x8 z8 = {0, 0, 0, 0, 0, 0, 0, 0};
      __builtin_amdgcn_tensor_load_to_lds(g0, g1, z4, z4, z8, 0);
      __builtin_amdgcn_s_wait_tensorcnt(0);
    }
#else
    {
      const int rr = tid >> 1, hh = (tid & 1) * 16;
      const bf16* ag = A + (size_t)(bm * 128 + rr) * K + k0 + hh;
      *(u32x4*)&atile[rr * 32 + hh]     = *(const u32x4*)ag;
      *(u32x4*)&atile[rr * 32 + hh + 8] = *(const u32x4*)(ag + 8);
    }
#endif
    {  // stage B^T: btile[n*BT_STRIDE + k] = B[k0+k][bn*64+n]
      const int kk = tid >> 3, n0 = (tid & 7) * 8;
      union { u32x4 q; bf16 h[8]; } u;
      u.q = *(const u32x4*)(Bm + (size_t)(k0 + kk) * N + bn * 64 + n0);
#pragma unroll
      for (int i = 0; i < 8; ++i) btile[(n0 + i) * BT_STRIDE + kk] = u.h[i];
    }
    __syncthreads();
    const v16bf af = load_frag_a16(atile + warp * 16 * 32, 32);
    const v16bf b0 = load_frag_bk(btile + 0 * 16 * BT_STRIDE, BT_STRIDE);
    const v16bf b1 = load_frag_bk(btile + 1 * 16 * BT_STRIDE, BT_STRIDE);
    const v16bf b2 = load_frag_bk(btile + 2 * 16 * BT_STRIDE, BT_STRIDE);
    const v16bf b3 = load_frag_bk(btile + 3 * 16 * BT_STRIDE, BT_STRIDE);
    acc0 = wmma_bf16(af, b0, acc0);
    acc1 = wmma_bf16(af, b1, acc1);
    acc2 = wmma_bf16(af, b2, acc2);
    acc3 = wmma_bf16(af, b3, acc3);
    __syncthreads();
  }

  const size_t crow = (size_t)bm * 128 + warp * 16;
#pragma unroll
  for (int r = 0; r < 8; ++r) {
    float* cp = C + (crow + r + 8 * half) * N + bn * 64 + nn;
    cp[0]  = acc0[r];
    cp[16] = acc1[r];
    cp[32] = acc2[r];
    cp[48] = acc3[r];
  }
}

// ---------------------------------------------------------------------------
// Epilogue: RoPE + RMSnorm for q/k, gate + ve for v. One wave per (b,t,slot).
// K written into padded [B,KV,TPAD,64] at row t+WINDOW.
// V written transposed into [B,KV,64,TPAD] at col t+WINDOW.
// ---------------------------------------------------------------------------
__global__ void qkv_epilogue_kernel(const float* __restrict__ qkvf,
                                    const float* __restrict__ x,
                                    const float* __restrict__ ve,
                                    const float* __restrict__ cosT,
                                    const float* __restrict__ sinT,
                                    const float* __restrict__ wgate,
                                    bf16* __restrict__ Qb,
                                    bf16* __restrict__ Kpad,
                                    bf16* __restrict__ Vt) {
  const int warp = threadIdx.x >> 5, lane = threadIdx.x & 31;
  const int wg = blockIdx.x * (blockDim.x >> 5) + warp;
  const int slot = wg % 24;
  const int bt = wg / 24;
  if (bt >= BB * TT) return;
  const int b = bt / TT, t = bt % TT;
  const float* base = qkvf + (size_t)bt * 1536;

  if (slot < 20) {
    const float* src;
    bf16* dst;
    if (slot < 16) {
      src = base + slot * 64;
      dst = Qb + (((size_t)(b * N_HEAD + slot)) * TT + t) * 64;
    } else {
      const int kv = slot - 16;
      src = base + 1024 + kv * 64;
      dst = Kpad + (((size_t)(b * N_KV + kv)) * TPAD + (WINDOW + t)) * 64;
    }
    float x1 = src[lane], x2 = src[lane + 32];
    float co = cosT[(size_t)t * 32 + lane];
    float si = sinT[(size_t)t * 32 + lane];
    float r1 = x1 * co - x2 * si;
    float r2 = x1 * si + x2 * co;
    float ss = r1 * r1 + r2 * r2;
#pragma unroll
    for (int m = 16; m >= 1; m >>= 1) ss += __shfl_xor(ss, m, 32);
    float inv = rsqrtf(ss * (1.0f / 64.0f) + 1e-6f);
    dst[lane]      = (bf16)(r1 * inv);
    dst[lane + 32] = (bf16)(r2 * inv);
  } else {
    const int kv = slot - 20;
    const float* src = base + 1280 + kv * 64;
    float g = x[(size_t)bt * N_EMBD + lane] * wgate[lane * N_KV + kv];
#pragma unroll
    for (int m = 16; m >= 1; m >>= 1) g += __shfl_xor(g, m, 32);
    g = 2.0f / (1.0f + __expf(-g));
    const float* vesrc = ve + (size_t)bt * (N_KV * 64) + kv * 64;
    bf16* dst = Vt + ((size_t)(b * N_KV + kv)) * 64 * TPAD + (WINDOW + t);
    dst[(size_t)lane * TPAD]        = (bf16)(src[lane]      + g * vesrc[lane]);
    dst[(size_t)(lane + 32) * TPAD] = (bf16)(src[lane + 32] + g * vesrc[lane + 32]);
  }
}

// ---------------------------------------------------------------------------
// Flash attention, sliding window 512, GQA rep=4. One wave per
// (b, head, 16-query tile); 17 uniform 32-key chunks; padded layouts make all
// fragment loads contiguous b128s with no address clamping.
// ---------------------------------------------------------------------------
__global__ void __launch_bounds__(256)
attn_kernel(const bf16* __restrict__ Qb, const bf16* __restrict__ Kpad,
            const bf16* __restrict__ Vt, bf16* __restrict__ Yb) {
  __shared__ bf16 plds[8 * 16 * 32];  // per-wave 16x32 P tile
  const int warp = threadIdx.x >> 5, lane = threadIdx.x & 31;
  const int wg = blockIdx.x * 8 + warp;
  const int nqt = TT / 16;
  const int qt = wg % nqt;
  const int h  = (wg / nqt) % N_HEAD;
  const int b  = wg / (nqt * N_HEAD);
  const int q0 = qt * 16;
  const int half = lane >> 4, nn = lane & 15;

  const bf16* Qp  = Qb   + (((size_t)(b * N_HEAD + h)) * TT + q0) * 64;
  const bf16* Kp  = Kpad + ((size_t)(b * N_KV + (h >> 2))) * TPAD * 64;
  const bf16* Vtp = Vt   + ((size_t)(b * N_KV + (h >> 2))) * 64 * TPAD;

  const v16bf aq0 = load_frag_a16(Qp, 64);       // dims 0..31
  const v16bf aq1 = load_frag_a16(Qp + 32, 64);  // dims 32..63

  float mrow[8], lrow[8];
  v8f o0 = {}, o1 = {}, o2 = {}, o3 = {};
#pragma unroll
  for (int r = 0; r < 8; ++r) { mrow[r] = -1e30f; lrow[r] = 0.0f; }

  bf16* myp = plds + warp * (16 * 32);

  for (int c = 0; c < 17; ++c) {
    const int t0 = q0 - WINDOW + c * 32;        // key chunk [t0, t0+32)
    const int row0 = t0 + WINDOW;               // padded row index (>= 0)

    // ---- S = (Q K^T) * scale ----
    v8f s[2];
#pragma unroll
    for (int ti = 0; ti < 2; ++ti) {
      const bf16* kbase = Kp + (size_t)(row0 + ti * 16) * 64;
      const v16bf bk0 = load_frag_bk(kbase, 64);       // dims 0..31
      const v16bf bk1 = load_frag_bk(kbase + 32, 64);  // dims 32..63
      v8f acc = {};
      acc = wmma_bf16(aq0, bk0, acc);
      acc = wmma_bf16(aq1, bk1, acc);
      s[ti] = acc;
    }

    // ---- mask + online softmax (rows live in one 16-lane half) ----
#pragma unroll
    for (int r = 0; r < 8; ++r) {
      const int i = q0 + r + 8 * half;
#pragma unroll
      for (int ti = 0; ti < 2; ++ti) {
        const int j = t0 + ti * 16 + nn;
        const float sv = s[ti][r] * 0.125f;  // 1/sqrt(64)
        const bool bad = (j > i) | (j < i - WINDOW) | (j < 0);
        s[ti][r] = bad ? -1e30f : sv;
      }
      float mx = fmaxf(s[0][r], s[1][r]);
#pragma unroll
      for (int msk = 8; msk >= 1; msk >>= 1)
        mx = fmaxf(mx, __shfl_xor(mx, msk, 32));
      const float mnew = fmaxf(mrow[r], mx);
      const float p0 = __expf(s[0][r] - mnew);
      const float p1 = __expf(s[1][r] - mnew);
      float rs = p0 + p1;
#pragma unroll
      for (int msk = 8; msk >= 1; msk >>= 1) rs += __shfl_xor(rs, msk, 32);
      const float alpha = __expf(mrow[r] - mnew);
      lrow[r] = lrow[r] * alpha + rs;
      mrow[r] = mnew;
      o0[r] = o0[r] * alpha;
      o1[r] = o1[r] * alpha;
      o2[r] = o2[r] * alpha;
      o3[r] = o3[r] * alpha;
      myp[(r + 8 * half) * 32 + nn]      = (bf16)p0;
      myp[(r + 8 * half) * 32 + 16 + nn] = (bf16)p1;
    }

    // ---- O += P @ V  (P from LDS as A-frag; V^T gives contiguous B-frags) ----
    const v16bf ap = load_frag_a16(myp, 32);
    const bf16* vbase = Vtp + (size_t)row0;
    const v16bf bv0 = load_frag_bk(vbase + (size_t)(0 * 16) * TPAD, TPAD);
    const v16bf bv1 = load_frag_bk(vbase + (size_t)(1 * 16) * TPAD, TPAD);
    const v16bf bv2 = load_frag_bk(vbase + (size_t)(2 * 16) * TPAD, TPAD);
    const v16bf bv3 = load_frag_bk(vbase + (size_t)(3 * 16) * TPAD, TPAD);
    o0 = wmma_bf16(ap, bv0, o0);
    o1 = wmma_bf16(ap, bv1, o1);
    o2 = wmma_bf16(ap, bv2, o2);
    o3 = wmma_bf16(ap, bv3, o3);
  }

  // ---- normalize + write Y[b, t, h*64 + d] as bf16 ----
#pragma unroll
  for (int r = 0; r < 8; ++r) {
    const int t = q0 + r + 8 * half;
    const float invl = 1.0f / lrow[r];
    const size_t rowoff = ((size_t)(b * TT + t)) * N_EMBD + h * 64;
    Yb[rowoff + 0  + nn] = (bf16)(o0[r] * invl);
    Yb[rowoff + 16 + nn] = (bf16)(o1[r] * invl);
    Yb[rowoff + 32 + nn] = (bf16)(o2[r] * invl);
    Yb[rowoff + 48 + nn] = (bf16)(o3[r] * invl);
  }
}

// ---------------------------------------------------------------------------
// Host launcher
// ---------------------------------------------------------------------------
extern "C" void kernel_launch(void* const* d_in, const int* in_sizes, int n_in,
                              void* d_out, int out_size, void* d_ws, size_t ws_size,
                              hipStream_t stream) {
  const float* x     = (const float*)d_in[0];
  const float* ve    = (const float*)d_in[1];
  const float* cosT  = (const float*)d_in[2];
  const float* sinT  = (const float*)d_in[3];
  const float* Wq    = (const float*)d_in[4];
  const float* Wk    = (const float*)d_in[5];
  const float* Wv    = (const float*)d_in[6];
  const float* Wproj = (const float*)d_in[7];
  const float* Wgate = (const float*)d_in[8];
  float* out = (float*)d_out;

  const size_t BTOT = (size_t)BB * TT;  // 4096
  char* ws = (char*)d_ws;
  bf16* xb    = (bf16*)ws;  ws += BTOT * N_EMBD * sizeof(bf16);
  bf16* wqkv  = (bf16*)ws;  ws += (size_t)N_EMBD * 1536 * sizeof(bf16);
  bf16* wproj = (bf16*)ws;  ws += (size_t)N_EMBD * N_EMBD * sizeof(bf16);
  float* qkvf = (float*)ws; ws += BTOT * 1536 * sizeof(float);
  bf16* Qb    = (bf16*)ws;  ws += BTOT * (N_HEAD * 64) * sizeof(bf16);
  bf16* Kpad  = (bf16*)ws;  ws += (size_t)BB * N_KV * TPAD * 64 * sizeof(bf16);
  bf16* Vt    = (bf16*)ws;  ws += (size_t)BB * N_KV * 64 * TPAD * sizeof(bf16);
  bf16* Yb    = (bf16*)ws;  ws += BTOT * N_EMBD * sizeof(bf16);

  // 1) conversions (vectorized x4)
  {
    int n4 = (int)(BTOT * N_EMBD / 4);
    f32_to_bf16_kernel<<<(n4 + 255) / 256, 256, 0, stream>>>(x, xb, n4);
  }
  {
    int n = N_EMBD * 1536;
    pack_wqkv_kernel<<<(n + 255) / 256, 256, 0, stream>>>(Wq, Wk, Wv, wqkv);
  }
  {
    int n4 = N_EMBD * N_EMBD / 4;
    f32_to_bf16_kernel<<<(n4 + 255) / 256, 256, 0, stream>>>(Wproj, wproj, n4);
  }

  // 2) QKV GEMM: [4096,1536] = [4096,1024] @ [1024,1536]
  {
    int blocks = (int)(BTOT / 128) * (1536 / 64);  // 32*24 = 768
    gemm_bf16_kernel<<<blocks, 256, 0, stream>>>(xb, wqkv, qkvf,
                                                 (int)BTOT, 1536, N_EMBD);
  }

  // 3) RoPE/RMSnorm/gate epilogue: BTOT*24 waves
  {
    int blocks = (int)(BTOT * 24 / 8);
    qkv_epilogue_kernel<<<blocks, 256, 0, stream>>>(qkvf, x, ve, cosT, sinT,
                                                    Wgate, Qb, Kpad, Vt);
  }

  // 4) attention: B*H*(T/16) waves = 4096
  {
    int waves = BB * N_HEAD * (TT / 16);
    attn_kernel<<<waves / 8, 256, 0, stream>>>(Qb, Kpad, Vt, Yb);
  }

  // 5) output projection: [4096,1024] = [4096,1024] @ [1024,1024]
  {
    int blocks = (int)(BTOT / 128) * (N_EMBD / 64);  // 32*16 = 512
    gemm_bf16_kernel<<<blocks, 256, 0, stream>>>(Yb, wproj, out,
                                                 (int)BTOT, N_EMBD, N_EMBD);
  }
  (void)in_sizes; (void)n_in; (void)out_size; (void)ws_size;
}
